// LocalPushPlus_48438641164510
// MI455X (gfx1250) — matrix operand order repaired and checked by reading, not compiled, not run on hardware
//
#include <hip/hip_runtime.h>

#define N_ 8192
#define D_ 512

#define BM 128
#define BN 256
#define BK 32
#define LDK 40   // padded LDS row stride in halves: 80B = 5*16B aligned, bank-conflict-free
#define CBLK 2   // column blocks handled per workgroup

typedef __attribute__((ext_vector_type(16))) _Float16 v16h;
typedef __attribute__((ext_vector_type(8)))  _Float16 v8h;
typedef __attribute__((ext_vector_type(8)))  float    v8f;

// ---- CDNA5 async global->LDS copy (ASYNCcnt-tracked, bypasses VGPRs) -------------
__device__ __forceinline__ void async_copy16(unsigned lds_addr, const void* gaddr) {
  asm volatile("global_load_async_to_lds_b128 %0, %1, off"
               :: "v"(lds_addr), "v"(gaddr) : "memory");
}
__device__ __forceinline__ void wait_async0() {
  asm volatile("s_wait_asynccnt 0x0" ::: "memory");
}

// ---------------- Kernel A: L2-normalize rows, pos distances, zero accumulators ----
__global__ __launch_bounds__(256) void normalize_pos_kernel(
    const float* __restrict__ features,
    const float* __restrict__ centers,
    const int*   __restrict__ labels,
    _Float16* __restrict__ fnorm,
    float* __restrict__ pos,
    float* __restrict__ rowsum,
    float* __restrict__ rowcnt)
{
  const int i = blockIdx.x;
  const int t = threadIdx.x;
  const int lab = labels[i];
  const float* frow = features + (size_t)i  * D_;
  const float* crow = centers  + (size_t)lab * D_;

  float f0 = frow[t], f1 = frow[t + 256];
  float c0 = crow[t], c1 = crow[t + 256];

  __shared__ float sF[256], sC[256], sD[256];
  sF[t] = f0 * f0 + f1 * f1;
  sC[t] = c0 * c0 + c1 * c1;
  sD[t] = f0 * c0 + f1 * c1;
  __syncthreads();
  for (int s = 128; s > 0; s >>= 1) {
    if (t < s) { sF[t] += sF[t + s]; sC[t] += sC[t + s]; sD[t] += sD[t + s]; }
    __syncthreads();
  }
  __shared__ float sInv;
  if (t == 0) {
    float nf = fmaxf(sqrtf(sF[0]), 1e-12f);
    float nc = fmaxf(sqrtf(sC[0]), 1e-12f);
    pos[i]    = 1.0f - sD[0] / (nf * nc);
    rowsum[i] = 0.0f;
    rowcnt[i] = 0.0f;
    sInv = 1.0f / nf;
  }
  __syncthreads();
  const float inv = sInv;
  fnorm[(size_t)i * D_ + t]       = (_Float16)(f0 * inv);
  fnorm[(size_t)i * D_ + t + 256] = (_Float16)(f1 * inv);
}

// ---------------- Kernel B: fused f16-WMMA similarity GEMM + masked row reduce -----
// Macro tile 128x256, 8 waves in 2x4 grid, each wave owns a 64x64 sub-tile
// (4x4 WMMA tiles -> 8 fragment loads per 16 WMMAs). Double-buffered LDS filled
// by async global->LDS copies; single barrier per K-step.
__global__ __launch_bounds__(256) void pairwise_kernel(
    const _Float16* __restrict__ fnorm,
    const float* __restrict__ pos,
    const int*   __restrict__ labels,
    float* __restrict__ rowsum,
    float* __restrict__ rowcnt)
{
  __shared__ _Float16 As[2][BM][LDK];   // 20,480 B
  __shared__ _Float16 Bs[2][BN][LDK];   // 40,960 B

  const int t      = threadIdx.x;
  const int lane   = t & 31;
  const int wid    = t >> 5;        // 0..7 waves
  const int wave_r = wid >> 2;      // 0..1 -> 64 rows each
  const int wave_c = wid & 3;       // 0..3 -> 64 cols each
  const int half1  = (lane >> 4) & 1;
  const int l16    = lane & 15;

  const int rowBase = blockIdx.x * BM;

  // Per-lane rows: i = rowBase + wave_r*64 + tm*16 + half1*8 + v  (C/D layout)
  float posR[4][8];
  int   labR[4][8];
#pragma unroll
  for (int tm = 0; tm < 4; ++tm)
#pragma unroll
    for (int v = 0; v < 8; ++v) {
      int i = rowBase + wave_r * 64 + tm * 16 + half1 * 8 + v;
      posR[tm][v] = pos[i];
      labR[tm][v] = labels[i];
    }

  float lsum[4][8], lcnt[4][8];
#pragma unroll
  for (int tm = 0; tm < 4; ++tm)
#pragma unroll
    for (int v = 0; v < 8; ++v) { lsum[tm][v] = 0.f; lcnt[tm][v] = 0.f; }

  for (int cbi = 0; cbi < CBLK; ++cbi) {
    const int colBase = (blockIdx.y * CBLK + cbi) * BN;

    int labJ[4];
#pragma unroll
    for (int tn = 0; tn < 4; ++tn)
      labJ[tn] = labels[colBase + wave_c * 64 + tn * 16 + l16];

    v8f acc[4][4];
#pragma unroll
    for (int tm = 0; tm < 4; ++tm)
#pragma unroll
      for (int tn = 0; tn < 4; ++tn) acc[tm][tn] = {};

    // issue async copies for one double-buffer stage (A: 2 chunks, B: 4 chunks / thread)
    auto issue = [&](int stg, int kbb) {
#pragma unroll
      for (int it = 0; it < 2; ++it) {
        int chunk = t + it * 256;              // 512 chunks: 128 rows x 4x16B
        int r = chunk >> 2, c8 = (chunk & 3) * 8;
        async_copy16((unsigned)(unsigned long long)&As[stg][r][c8],
                     fnorm + (size_t)(rowBase + r) * D_ + kbb + c8);
      }
#pragma unroll
      for (int it = 0; it < 4; ++it) {
        int chunk = t + it * 256;              // 1024 chunks: 256 rows x 4x16B
        int r = chunk >> 2, c8 = (chunk & 3) * 8;
        async_copy16((unsigned)(unsigned long long)&Bs[stg][r][c8],
                     fnorm + (size_t)(colBase + r) * D_ + kbb + c8);
      }
    };

    issue(0, 0);
    int s = 0;
    for (int kb = 0; kb < D_; kb += BK, s ^= 1) {
      wait_async0();       // this wave's stage-s copies have landed in LDS
      __syncthreads();     // everyone's copies visible; prior reads of s^1 done
      if (kb + BK < D_) issue(s ^ 1, kb + BK);   // prefetch next stage under compute

      // A fragments (16x32 f16): lanes 0-15: K {0-7,16-23}; lanes 16-31: K {8-15,24-31}
      v16h afrag[4];
#pragma unroll
      for (int tm = 0; tm < 4; ++tm) {
        int r = wave_r * 64 + tm * 16 + l16;
        v8h alo = *reinterpret_cast<const v8h*>(&As[s][r][half1 * 8]);
        v8h ahi = *reinterpret_cast<const v8h*>(&As[s][r][16 + half1 * 8]);
        afrag[tm] = __builtin_shufflevector(alo, ahi,
            0, 1, 2, 3, 4, 5, 6, 7, 8, 9, 10, 11, 12, 13, 14, 15);
      }
      // B fragments (32x16 f16): lanes 0-15: K {0-7,16-23}; lanes 16-31: K {8-15,24-31}
      v16h bfrag[4];
#pragma unroll
      for (int tn = 0; tn < 4; ++tn) {
        int c = wave_c * 64 + tn * 16 + l16;
        v8h blo = *reinterpret_cast<const v8h*>(&Bs[s][c][half1 * 8]);
        v8h bhi = *reinterpret_cast<const v8h*>(&Bs[s][c][16 + half1 * 8]);
        bfrag[tn] = __builtin_shufflevector(blo, bhi,
            0, 1, 2, 3, 4, 5, 6, 7, 8, 9, 10, 11, 12, 13, 14, 15);
      }
#pragma unroll
      for (int tm = 0; tm < 4; ++tm)
#pragma unroll
        for (int tn = 0; tn < 4; ++tn)
          acc[tm][tn] = __builtin_amdgcn_wmma_f32_16x16x32_f16(
              false, afrag[tm], false, bfrag[tn],
              (short)0, acc[tm][tn], false, false);
    }

    // Epilogue: losses_ij = pos_i + sim_ij - 1; valid if labels differ and loss>0
#pragma unroll
    for (int tm = 0; tm < 4; ++tm) {
#pragma unroll
      for (int tn = 0; tn < 4; ++tn) {
#pragma unroll
        for (int v = 0; v < 8; ++v) {
          float sv = acc[tm][tn][v];
          float lossv = posR[tm][v] + sv - 1.0f;
          bool valid = (labR[tm][v] != labJ[tn]) && (lossv > 0.0f);
          if (valid) { lsum[tm][v] += lossv; lcnt[tm][v] += 1.0f; }
        }
      }
    }
    __syncthreads();   // finish reads of last stage before next cbi reuses buffers
  }

  // Reduce over the 16 lanes of each half-wave holding the same row, then atomics
#pragma unroll
  for (int tm = 0; tm < 4; ++tm) {
#pragma unroll
    for (int v = 0; v < 8; ++v) {
      float sv = lsum[tm][v];
      float cv = lcnt[tm][v];
#pragma unroll
      for (int m = 1; m < 16; m <<= 1) {
        sv += __shfl_xor(sv, m, 32);
        cv += __shfl_xor(cv, m, 32);
      }
      if (l16 == 0) {
        int i = rowBase + wave_r * 64 + tm * 16 + half1 * 8 + v;
        atomicAdd(&rowsum[i], sv);
        atomicAdd(&rowcnt[i], cv);
      }
    }
  }
}

// ---------------- Kernel C: finalize scalar loss ----------------------------------
__global__ __launch_bounds__(256) void finalize_kernel(
    const float* __restrict__ rowsum,
    const float* __restrict__ rowcnt,
    float* __restrict__ out)
{
  const int t = threadIdx.x;
  float sumLoss = 0.f, nValid = 0.f;
  for (int i = t; i < N_; i += 256) {
    float cnt = rowcnt[i];
    float sl  = rowsum[i] / fmaxf(cnt, 1.0f);
    sumLoss += sl;
    if (sl > 0.f) nValid += 1.f;
  }
  __shared__ float sS[256], sV[256];
  sS[t] = sumLoss; sV[t] = nValid;
  __syncthreads();
  for (int s = 128; s > 0; s >>= 1) {
    if (t < s) { sS[t] += sS[t + s]; sV[t] += sV[t + s]; }
    __syncthreads();
  }
  if (t == 0) {
    float nv = sV[0];
    out[0] = (nv > 0.f) ? (sS[0] / nv) : sS[0];
  }
}

extern "C" void kernel_launch(void* const* d_in, const int* in_sizes, int n_in,
                              void* d_out, int out_size, void* d_ws, size_t ws_size,
                              hipStream_t stream) {
  const float* features = (const float*)d_in[0];
  const float* centers  = (const float*)d_in[1];
  const int*   labels   = (const int*)d_in[2];
  float* out = (float*)d_out;

  char* ws = (char*)d_ws;
  _Float16* fnorm = (_Float16*)ws;                                 // 8 MB
  float* pos    = (float*)(ws + (size_t)N_ * D_ * sizeof(_Float16));
  float* rowsum = pos + N_;
  float* rowcnt = rowsum + N_;

  normalize_pos_kernel<<<N_, 256, 0, stream>>>(features, centers, labels,
                                               fnorm, pos, rowsum, rowcnt);
  dim3 gridB(N_ / BM, N_ / (BN * CBLK));
  pairwise_kernel<<<gridB, 256, 0, stream>>>(fnorm, pos, labels, rowsum, rowcnt);
  finalize_kernel<<<1, 256, 0, stream>>>(rowsum, rowcnt, out);
}
